// MultiHeadAttention_68350109548803
// MI455X (gfx1250) — compile-verified, matrix operand order checked by reading, not run on hardware
//
#include <hip/hip_runtime.h>
#include <hip/hip_bf16.h>
#include <stdint.h>

// MI455X / gfx1250, wave32. GEMMs + attention on v_wmma_f32_16x16x32_f16,
// all LDS staging via CDNA5 async-to-LDS DMA (ASYNCcnt) with double buffering.
// Attention exploits sorted doc_ids: skips fully-masked key blocks (~8x less
// work) and uses a block-uniform fast path that drops per-element masking on
// interior tiles.

typedef __attribute__((ext_vector_type(16))) _Float16 v16h;
typedef __attribute__((ext_vector_type(8)))  _Float16 v8h;
typedef __attribute__((ext_vector_type(8)))  float    v8f;

union AFrag { v16h v; v8h h[2]; };

__device__ __forceinline__ v8f wmma16(v16h a, v16h b, v8f c) {
  return __builtin_amdgcn_wmma_f32_16x16x32_f16(false, a, false, b, (short)0, c,
                                                false, false);
}

// Async global->LDS copy of 16 bytes (per-lane addresses), ASYNCcnt-tracked.
__device__ __forceinline__ void async_ld_b128(uint32_t lds_off, const void* g) {
  asm volatile("global_load_async_to_lds_b128 %0, %1, off"
               :: "v"(lds_off), "v"((unsigned long long)(uintptr_t)g)
               : "memory");
}
__device__ __forceinline__ void wait_async0() {
  asm volatile("s_wait_asynccnt 0x0" ::: "memory");
}
__device__ __forceinline__ uint32_t lds_off_of(const void* p) {
  return (uint32_t)(uintptr_t)p;  // low 32 bits of generic shared addr = LDS offset
}

// ---------------------------------------------------------------------------
// Elementwise f32 -> f16
// ---------------------------------------------------------------------------
__global__ __launch_bounds__(256) void cvt_f16_kernel(
    const float* __restrict__ in, _Float16* __restrict__ out, int n) {
  int i = blockIdx.x * 256 + threadIdx.x;
  if (i < n) out[i] = (_Float16)in[i];
}

// ---------------------------------------------------------------------------
// out[c][r] = (f16) in[r][c]; 32x32 LDS tiles. R%32==0, C%32==0.
// ---------------------------------------------------------------------------
template <typename Tin>
__global__ __launch_bounds__(256) void transpose_cvt_kernel(
    const Tin* __restrict__ in, _Float16* __restrict__ out, int R, int C,
    int ldin, int ldout) {
  __shared__ float tile[32][33];
  const int r0 = blockIdx.y * 32, c0 = blockIdx.x * 32;
  const int tr = threadIdx.x >> 5, tc = threadIdx.x & 31;
#pragma unroll
  for (int i = 0; i < 4; ++i) {
    int r = tr + i * 8;
    tile[r][tc] = (float)in[(size_t)(r0 + r) * ldin + c0 + tc];
  }
  __syncthreads();
#pragma unroll
  for (int i = 0; i < 4; ++i) {
    int r = tr + i * 8;
    out[(size_t)(c0 + r) * ldout + r0 + tc] = (_Float16)tile[tc][r];
  }
}

// ---------------------------------------------------------------------------
// C[M,N] = A[M,K] * B[K,N] with A f16 row-major [M][K], Bt f16 [N][K].
// 128x128 block tile, K-step 32, 8 waves (2x4) x (64x32) sub-tiles.
// Tiles staged via async-to-LDS, double buffered.
// ---------------------------------------------------------------------------
template <typename TC>
__global__ __launch_bounds__(256) void gemm_wmma_kernel(
    const _Float16* __restrict__ A, const _Float16* __restrict__ Bt,
    TC* __restrict__ C, int M, int N, int K) {
  __shared__ _Float16 As[2][128][40];  // 80B rows: 16B-aligned chunks
  __shared__ _Float16 Bs[2][128][40];

  const int tid = threadIdx.x;
  const int lane = tid & 31;
  const int wid = tid >> 5;
  const int wm = wid >> 2;  // 0..1
  const int wn = wid & 3;   // 0..3
  const int mblk = blockIdx.y * 128;
  const int nblk = blockIdx.x * 128;
  const int half8 = (lane & 16) ? 8 : 0;
  const int l16 = lane & 15;

  const uint32_t asb = lds_off_of(&As[0][0][0]);
  const uint32_t bsb = lds_off_of(&Bs[0][0][0]);
  const uint32_t bstride = 128 * 40 * 2;

  auto stage = [&](int buf, int k0) {
#pragma unroll
    for (int i = 0; i < 2; ++i) {
      int idx = tid + i * 256;  // 512 chunks of 16B per tile
      int row = idx >> 2, c4 = idx & 3;
      uint32_t loff = (uint32_t)(row * 80 + c4 * 16) + buf * bstride;
      async_ld_b128(asb + loff, A + (size_t)(mblk + row) * K + k0 + c4 * 8);
      async_ld_b128(bsb + loff, Bt + (size_t)(nblk + row) * K + k0 + c4 * 8);
    }
  };

  v8f acc[4][2];
#pragma unroll
  for (int mt = 0; mt < 4; ++mt)
#pragma unroll
    for (int nt = 0; nt < 2; ++nt) acc[mt][nt] = {};

  stage(0, 0);
  const int nk = K >> 5;
  for (int i = 0; i < nk; ++i) {
    wait_async0();
    __syncthreads();  // tiles for buf published
    if (i + 1 < nk) stage((i + 1) & 1, (i + 1) * 32);
    const int buf = i & 1;

    AFrag a[4], bf[2];
#pragma unroll
    for (int mt = 0; mt < 4; ++mt) {
      int r = wm * 64 + mt * 16 + l16;
      a[mt].h[0] = *(const v8h*)&As[buf][r][half8];
      a[mt].h[1] = *(const v8h*)&As[buf][r][half8 + 16];
    }
#pragma unroll
    for (int nt = 0; nt < 2; ++nt) {
      int c = wn * 32 + nt * 16 + l16;
      bf[nt].h[0] = *(const v8h*)&Bs[buf][c][half8];
      bf[nt].h[1] = *(const v8h*)&Bs[buf][c][half8 + 16];
    }
#pragma unroll
    for (int mt = 0; mt < 4; ++mt)
#pragma unroll
      for (int nt = 0; nt < 2; ++nt)
        acc[mt][nt] = wmma16(a[mt].v, bf[nt].v, acc[mt][nt]);
  }

#pragma unroll
  for (int mt = 0; mt < 4; ++mt)
#pragma unroll
    for (int nt = 0; nt < 2; ++nt)
#pragma unroll
      for (int r = 0; r < 8; ++r) {
        int gr = mblk + wm * 64 + mt * 16 + r + half8;
        int gc = nblk + wn * 32 + nt * 16 + l16;
        C[(size_t)gr * N + gc] = (TC)acc[mt][nt][r];
      }
}

// ---------------------------------------------------------------------------
// RoPE on Q and K halves of f16 qkv [B*T,6144], in place; pair-per-thread.
// ---------------------------------------------------------------------------
__global__ __launch_bounds__(256) void rope_kernel(
    _Float16* __restrict__ qkv, const float* __restrict__ sinp,
    const float* __restrict__ cosp) {
  int gid = blockIdx.x * blockDim.x + threadIdx.x;
  int d1 = gid & 63;
  int h = (gid >> 6) & 15;
  int t = (gid >> 10) & 2047;
  int bs = gid >> 21;  // sel*2 + b  (B==2)
  int b = bs & 1;
  int sel = bs >> 1;   // 0 = Q, 1 = K

  size_t base = (size_t)(b * 2048 + t) * 6144 + sel * 2048 + h * 128;
  float x1 = (float)qkv[base + d1];
  float x2 = (float)qkv[base + d1 + 64];
  float s1 = sinp[t * 128 + d1],      c1 = cosp[t * 128 + d1];
  float s2 = sinp[t * 128 + d1 + 64], c2 = cosp[t * 128 + d1 + 64];
  qkv[base + d1]      = (_Float16)(x1 * c1 - x2 * s1);
  qkv[base + d1 + 64] = (_Float16)(x2 * c2 + x1 * s2);
}

// ---------------------------------------------------------------------------
// Flash attention: block = 64 q rows x 1 head; 4 waves x 16 rows.
// doc_ids are sorted -> key blocks wholly before this q-block's first document
// are fully masked and skipped entirely; interior tiles skip per-element
// masking via a block-uniform test. Double-buffered async K/V staging.
// ---------------------------------------------------------------------------
#define AT_T 2048
#define QKVN 6144
#define ATT_SCALE 0.08838834764831845f  // 1/sqrt(128)

__global__ __launch_bounds__(128) void attn_kernel(
    const _Float16* __restrict__ qkv, const _Float16* __restrict__ vT,
    const int* __restrict__ doc, _Float16* __restrict__ attn_out) {
  __shared__ _Float16 Qs[64][136];
  __shared__ _Float16 Ks[2][64][136];
  __shared__ _Float16 Vt[2][128][72];  // Vt[d][key]
  __shared__ _Float16 Ps[4][16][72];

  const int tid = threadIdx.x;
  const int lane = tid & 31;
  const int wid = tid >> 5;
  const int qb = blockIdx.x * 64;
  const int h = blockIdx.y;
  const int b = blockIdx.z;
  const int half8 = (lane & 16) ? 8 : 0;
  const int l16 = lane & 15;
  const size_t rowbase = (size_t)b * AT_T;

  const uint32_t qsb = lds_off_of(&Qs[0][0]);
  const uint32_t ksb = lds_off_of(&Ks[0][0][0]);
  const uint32_t vtb = lds_off_of(&Vt[0][0][0]);
  const uint32_t kstride = 64 * 136 * 2;
  const uint32_t vstride = 128 * 72 * 2;

  // First key block that can contain this q-block's documents: binary-search
  // the sorted doc array for the first t with doc[t] == doc[qb]. All earlier
  // key blocks are fully masked (their docs are strictly smaller).
  int jb_start;
  {
    const int target = doc[rowbase + qb];
    int lo = 0, hi = qb;
    while (lo < hi) {
      int mid = (lo + hi) >> 1;
      if (doc[rowbase + mid] < target) lo = mid + 1; else hi = mid;
    }
    jb_start = lo & ~63;
  }
  const int docQhi = doc[rowbase + qb + 63];

  auto stageKV = [&](int buf, int jb) {
#pragma unroll
    for (int i = 0; i < 8; ++i) {  // K: 64 rows x 16 chunks
      int idx = tid + i * 128;
      int row = idx >> 4, c8 = idx & 15;
      async_ld_b128(ksb + buf * kstride + (uint32_t)(row * 272 + c8 * 16),
                    qkv + (rowbase + jb + row) * QKVN + 2048 + h * 128 + c8 * 8);
    }
#pragma unroll
    for (int i = 0; i < 8; ++i) {  // V^T: 128 d-rows x 8 chunks
      int idx = tid + i * 128;
      int drow = idx >> 3, c8 = idx & 7;
      async_ld_b128(vtb + buf * vstride + (uint32_t)(drow * 144 + c8 * 16),
                    vT + (size_t)(h * 128 + drow) * 4096 + b * 2048 + jb + c8 * 8);
    }
  };

  // Stage Q + first useful K/V block
#pragma unroll
  for (int i = 0; i < 8; ++i) {
    int idx = tid + i * 128;
    int row = idx >> 4, c8 = idx & 15;
    async_ld_b128(qsb + (uint32_t)(row * 272 + c8 * 16),
                  qkv + (rowbase + qb + row) * QKVN + h * 128 + c8 * 8);
  }
  stageKV(0, jb_start);

  int dq[8];
#pragma unroll
  for (int r = 0; r < 8; ++r)
    dq[r] = doc[rowbase + qb + wid * 16 + r + half8];

  wait_async0();
  __syncthreads();

  AFrag aq[4];
  {
    int qrow = wid * 16 + l16;
#pragma unroll
    for (int c = 0; c < 4; ++c) {
      aq[c].h[0] = *(const v8h*)&Qs[qrow][c * 32 + half8];
      aq[c].h[1] = *(const v8h*)&Qs[qrow][c * 32 + half8 + 16];
    }
  }

  float mrun[8], lrun[8];
  v8f o[8];
#pragma unroll
  for (int r = 0; r < 8; ++r) { mrun[r] = -3.0e38f; lrun[r] = 0.f; }
#pragma unroll
  for (int dt = 0; dt < 8; ++dt) o[dt] = {};

  int buf = 0;
  for (int jb = jb_start; jb <= qb; jb += 64) {
    // S = Q K^T * scale  (mask applied below only when needed)
    v8f s[4];
#pragma unroll
    for (int jn = 0; jn < 4; ++jn) {
      int krow = jn * 16 + l16;
      v8f accs = {};
#pragma unroll
      for (int c = 0; c < 4; ++c) {
        AFrag bk;
        bk.h[0] = *(const v8h*)&Ks[buf][krow][c * 32 + half8];
        bk.h[1] = *(const v8h*)&Ks[buf][krow][c * 32 + half8 + 16];
        accs = wmma16(aq[c].v, bk.v, accs);
      }
#pragma unroll
      for (int r = 0; r < 8; ++r) s[jn][r] = accs[r] * ATT_SCALE;
    }

    // Block-uniform: tile is fully unmasked iff strictly below the diagonal
    // and one document covers cols jb..jb+63 and rows qb..qb+63.
    const bool fullTile = (jb + 64 <= qb) && (doc[rowbase + jb] == docQhi);
    if (!fullTile) {
#pragma unroll
      for (int jn = 0; jn < 4; ++jn) {
        int colg = jb + jn * 16 + l16;
        int dkc = doc[rowbase + jb + jn * 16 + l16];
#pragma unroll
        for (int r = 0; r < 8; ++r) {
          int rowg = qb + wid * 16 + r + half8;
          bool ok = (colg <= rowg) && (dkc == dq[r]);
          if (!ok) s[jn][r] = -1.0e30f;
        }
      }
    }

    // Online softmax (row reductions within 16-lane groups)
    float mx[8];
#pragma unroll
    for (int r = 0; r < 8; ++r)
      mx[r] = fmaxf(fmaxf(s[0][r], s[1][r]), fmaxf(s[2][r], s[3][r]));
#pragma unroll
    for (int off = 1; off < 16; off <<= 1)
#pragma unroll
      for (int r = 0; r < 8; ++r)
        mx[r] = fmaxf(mx[r], __shfl_xor(mx[r], off, 32));

    float alpha[8];
#pragma unroll
    for (int r = 0; r < 8; ++r) {
      float mnew = fmaxf(mrun[r], mx[r]);
      alpha[r] = __expf(mrun[r] - mnew);
      mrun[r] = mnew;
    }
#pragma unroll
    for (int jn = 0; jn < 4; ++jn)
#pragma unroll
      for (int r = 0; r < 8; ++r) s[jn][r] = __expf(s[jn][r] - mrun[r]);

    float ls[8];
#pragma unroll
    for (int r = 0; r < 8; ++r)
      ls[r] = (s[0][r] + s[1][r]) + (s[2][r] + s[3][r]);
#pragma unroll
    for (int off = 1; off < 16; off <<= 1)
#pragma unroll
      for (int r = 0; r < 8; ++r) ls[r] += __shfl_xor(ls[r], off, 32);
#pragma unroll
    for (int r = 0; r < 8; ++r) lrun[r] = lrun[r] * alpha[r] + ls[r];
#pragma unroll
    for (int dt = 0; dt < 8; ++dt)
#pragma unroll
      for (int r = 0; r < 8; ++r) o[dt][r] *= alpha[r];

    // P: C-fragment -> per-wave LDS patch -> A-fragment
#pragma unroll
    for (int jn = 0; jn < 4; ++jn)
#pragma unroll
      for (int r = 0; r < 8; ++r)
        Ps[wid][r + half8][jn * 16 + l16] = (_Float16)s[jn][r];
    __syncthreads();  // mid barrier: Ks reads + Ps writes done everywhere

    if (jb + 64 <= qb) stageKV(buf ^ 1, jb + 64);  // overlap DMA with P.V

    AFrag ap[2];
#pragma unroll
    for (int c = 0; c < 2; ++c) {
      ap[c].h[0] = *(const v8h*)&Ps[wid][l16][c * 32 + half8];
      ap[c].h[1] = *(const v8h*)&Ps[wid][l16][c * 32 + half8 + 16];
    }
#pragma unroll
    for (int dt = 0; dt < 8; ++dt) {
      int vr = dt * 16 + l16;
      AFrag bv0, bv1;
      bv0.h[0] = *(const v8h*)&Vt[buf][vr][half8];
      bv0.h[1] = *(const v8h*)&Vt[buf][vr][half8 + 16];
      bv1.h[0] = *(const v8h*)&Vt[buf][vr][32 + half8];
      bv1.h[1] = *(const v8h*)&Vt[buf][vr][32 + half8 + 16];
      o[dt] = wmma16(ap[0].v, bv0.v, o[dt]);
      o[dt] = wmma16(ap[1].v, bv1.v, o[dt]);
    }

    if (jb + 64 <= qb) {
      wait_async0();
      __syncthreads();  // next tiles published; all P.V reads done
      buf ^= 1;
    }
  }

#pragma unroll
  for (int r = 0; r < 8; ++r) lrun[r] = 1.0f / lrun[r];
#pragma unroll
  for (int dt = 0; dt < 8; ++dt)
#pragma unroll
    for (int r = 0; r < 8; ++r) {
      int gr = qb + wid * 16 + r + half8;
      int gc = h * 128 + dt * 16 + l16;
      attn_out[((size_t)(b * AT_T + gr)) * 2048 + gc] =
          (_Float16)(o[dt][r] * lrun[r]);
    }
}

// ---------------------------------------------------------------------------
extern "C" void kernel_launch(void* const* d_in, const int* in_sizes, int n_in,
                              void* d_out, int out_size, void* d_ws,
                              size_t ws_size, hipStream_t stream) {
  (void)in_sizes; (void)n_in; (void)out_size; (void)ws_size;
  const float* x    = (const float*)d_in[0];  // [2,2048,2048]
  const float* sinp = (const float*)d_in[1];  // [2048,128]
  const float* cosp = (const float*)d_in[2];  // [2048,128]
  const float* Wqkv = (const float*)d_in[3];  // [2048,6144]
  const float* Wout = (const float*)d_in[4];  // [2048,2048]
  const int*   doc  = (const int*)d_in[5];    // [2,2048]
  float* out = (float*)d_out;                 // [2,2048,2048]

  const int M = 4096;  // B*T
  char* ws = (char*)d_ws;
  _Float16* xh    = (_Float16*)ws;                       // 16 MiB
  _Float16* WqkvT = (_Float16*)(ws + (size_t)16777216);  // 24 MiB
  _Float16* WoutT = (_Float16*)(ws + (size_t)41943040);  //  8 MiB
  _Float16* qkv   = (_Float16*)(ws + (size_t)50331648);  // 48 MiB
  // Aliases: dead-after-GEMM1 regions reused.
  _Float16* vT    = WqkvT;  // [2048 = h*128+d][4096 = b*T+t]
  _Float16* attnh = xh;     // [4096][2048] f16

  // 0) Pre-pass: f16 copies / transposes (one-time, bandwidth-trivial)
  cvt_f16_kernel<<<(M * 2048) / 256, 256, 0, stream>>>(x, xh, M * 2048);
  transpose_cvt_kernel<float><<<dim3(6144 / 32, 2048 / 32), 256, 0, stream>>>(
      Wqkv, WqkvT, 2048, 6144, 6144, 2048);
  transpose_cvt_kernel<float><<<dim3(2048 / 32, 2048 / 32), 256, 0, stream>>>(
      Wout, WoutT, 2048, 2048, 2048, 2048);

  // 1) qkv = x @ W_qkv  (f16 x f16 -> f32 acc -> f16)
  gemm_wmma_kernel<_Float16><<<dim3(6144 / 128, M / 128), 256, 0, stream>>>(
      xh, WqkvT, qkv, M, 6144, 2048);

  // 2) RoPE on Q,K in place
  rope_kernel<<<(2 * 2 * 2048 * 16 * 64) / 256, 256, 0, stream>>>(qkv, sinp,
                                                                  cosp);

  // 3) Transpose V region: vT[h*128+d][b*T+t] (reuses WqkvT space)
  transpose_cvt_kernel<_Float16><<<dim3(2048 / 32, M / 32), 256, 0, stream>>>(
      qkv + 4096, vT, M, 2048, 6144, M);

  // 4) Flash attention -> attnh [4096,2048] f16 (reuses xh space)
  attn_kernel<<<dim3(2048 / 64, 16, 2), 128, 0, stream>>>(qkv, vT, doc, attnh);

  // 5) out = attn @ W_out (f32 result)
  gemm_wmma_kernel<float><<<dim3(2048 / 128, M / 128), 256, 0, stream>>>(
      attnh, WoutT, out, M, 2048, 2048);
}
